// MyRnn_38663295599192
// MI455X (gfx1250) — compile-verified, matrix operand order checked by reading, not run on hardware
//
#include <hip/hip_runtime.h>
#include <hip/hip_bf16.h>

// ---------------------------------------------------------------------------
// 2-layer LSTM (B=128, T=80, E=100, U=2048) for MI455X / gfx1250.
//  * Recurrent weights converted + pre-swizzled once per launch into the bf16
//    WMMA B-fragment layout (~103MB bf16 -> fully L2-resident on 192MB L2,
//    vs 205MB fp32 which would spill to HBM every timestep), with a
//    gate-grouping column permutation so each workgroup owns {i,f,g,o} of a
//    16-unit slice and can run the LSTM gate epilogue entirely in-workgroup.
//  * Per timestep, per layer: one fused GEMM+gate kernel,
//    z = [x|h] @ [W;U] via v_wmma_f32_16x16x32_bf16.  A (activations) is
//    staged into LDS with GLOBAL_LOAD_ASYNC_TO_LDS_B128 (ASYNCcnt-tracked,
//    no VGPR round-trip), double buffered over 64-wide K-chunks; B (weights)
//    streams from pre-swizzled global with global_prefetch ahead.
//  * h state ping-pongs between two bf16 buffers (cross-WG WAR safety);
//    c state stays fp32 in-place.
// Workspace requirement: ~112 MB (see offsets in kernel_launch).
// ---------------------------------------------------------------------------

typedef __attribute__((ext_vector_type(16))) __bf16 v16bf;
typedef __attribute__((ext_vector_type(8)))  float  v8f;

union FragU { uint4 q[2]; v16bf v; };

#define BATCH 128
#define SEQT  80
#define VOCAB 10000
#define EMB   100
#define EPAD  128
#define UNITS 2048
#define NCOLS 8192            // 4*UNITS
#define NTILES 512            // NCOLS/16
#define KC0   68              // (EPAD + UNITS)/32  = 2176/32  (32-granular)
#define KC1   128             // (UNITS + UNITS)/32 = 4096/32

static __device__ __forceinline__ unsigned short f2bf(float f) {
    unsigned u = __float_as_uint(f);
    u += 0x7FFFu + ((u >> 16) & 1u);       // round-to-nearest-even
    return (unsigned short)(u >> 16);
}
static __device__ __forceinline__ float sigm(float x) {
    return 1.0f / (1.0f + __expf(-x));
}

// --------------------------- zero-init state -------------------------------
__global__ __launch_bounds__(256) void k_zero(unsigned int* p, size_t n) {
    size_t i = (size_t)blockIdx.x * 256 + threadIdx.x;
    if (i < n) p[i] = 0u;
}

// ---------------- embedding gather -> padded bf16 X  [T][B][128] -----------
__global__ __launch_bounds__(256) void k_embed(const int* __restrict__ tokens,
                                               const float* __restrict__ emb,
                                               unsigned short* __restrict__ xg) {
    size_t idx = (size_t)blockIdx.x * 256 + threadIdx.x;   // T*B*128 = 1310720
    if (idx >= (size_t)SEQT * BATCH * EPAD) return;
    int e = (int)(idx & 127);
    int b = (int)((idx >> 7) & 127);
    int t = (int)(idx >> 14);
    float v = 0.0f;
    if (e < EMB) {
        int tok = tokens[b * SEQT + t];
        v = emb[(size_t)tok * EMB + e];
    }
    xg[idx] = f2bf(v);
}

// ------ weight convert + swizzle into WMMA B-fragment layout ---------------
// B fragment layout (mirror of 16-bit A layout, ISA 7.12.2):
//   lane l (0..31): column n = tile*16 + (l&15)
//                   K values: base = kc*32 + (l>>4)*8 ; e<8 -> base+e,
//                                                        e>=8 -> base+8+e
// Column permutation: global col j = wg*64 + 16*gate + ul  maps to original
// column 2048*gate + (wg*16 + ul)  (so one WG owns all 4 gates of 16 units).
// Row k: k<K0real from S0, K0real<=k<K0pad -> 0, else S1[k-K0pad].
__global__ __launch_bounds__(256) void k_swizzle(const float* __restrict__ S0,
                                                 int K0real, int K0pad,
                                                 const float* __restrict__ S1,
                                                 unsigned short* __restrict__ dst,
                                                 int KC) {
    size_t idx = (size_t)blockIdx.x * 256 + threadIdx.x;
    size_t total = (size_t)NTILES * KC * 512;
    if (idx >= total) return;
    int e    = (int)(idx & 15);
    int lane = (int)((idx >> 4) & 31);
    size_t rem = idx >> 9;
    int kc = (int)(rem % KC);
    int nt = (int)(rem / KC);
    int j  = nt * 16 + (lane & 15);
    int k  = kc * 32 + ((lane >> 4) << 3) + (e < 8 ? e : 8 + e);
    int wg = j >> 6, jl = j & 63, gate = jl >> 4, ul = jl & 15;
    int oc = (gate << 11) + wg * 16 + ul;                 // original column
    float v;
    if (k < K0real)      v = S0[(size_t)k * NCOLS + oc];
    else if (k < K0pad)  v = 0.0f;
    else                 v = S1[(size_t)(k - K0pad) * NCOLS + oc];
    dst[idx] = f2bf(v);
}

// ------------------- fused GEMM + LSTM gate step kernel --------------------
// grid = 128 WGs (each: 16 units x 4 gates = 64 cols), block = 256 (8 waves)
// wave w: ntile = w&3 (16 cols), mhalf = w>>2 (rows 64*mhalf..+63) -> 4 tiles
// K streamed in 64-wide chunks (8 WMMAs per chunk), A double-buffered in LDS
// via async-to-LDS copies, one workgroup barrier per 64-chunk.
__global__ __launch_bounds__(256) void k_lstm_step(
        const unsigned short* __restrict__ Bw, int KC32,
        const unsigned short* __restrict__ A0, int s0, int splitK,
        const unsigned short* __restrict__ A1, int s1,
        const float* __restrict__ bias,
        float* __restrict__ c,
        unsigned short* __restrict__ hout,
        float* __restrict__ hf) {
    __shared__ unsigned short As[2][BATCH][64];  // 32 KB, double buffered
    __shared__ float Zs[BATCH][68];              // 34 KB z tile (+pad)

    const int tid   = threadIdx.x;
    const int wg    = blockIdx.x;
    const int lane  = tid & 31;
    const int wave  = tid >> 5;
    const int ntile = wave & 3;
    const int mhalf = wave >> 2;
    const int KCW   = KC32 >> 1;                 // number of 64-wide chunks

    const unsigned short* bw =
        Bw + ((size_t)(wg * 4 + ntile) * KC32) * 512 + lane * 16;

    v8f acc[4];
#pragma unroll
    for (int i = 0; i < 4; ++i)
#pragma unroll
        for (int v = 0; v < 8; ++v) acc[i][v] = 0.0f;

    // Async stage of one 64-wide K chunk of A into LDS buffer `buf`.
    // 256 threads x 64B = 16KB.  Each 32-ushort segment lies entirely on one
    // side of splitK (splitK is a multiple of 32).
    auto stage = [&](int kcw, int buf) {
        int r = tid >> 1, seg = tid & 1;
        int kg = kcw * 64 + seg * 32;
        const unsigned short* s = (kg < splitK)
            ? (A0 + (size_t)r * s0 + kg)
            : (A1 + (size_t)r * s1 + (kg - splitK));
        // LDS offset = low 32 bits of the generic address (ISA: LDS aperture
        // places the offset in addr[31:0]).
        unsigned dst = (unsigned)(uintptr_t)(void*)&As[buf][r][seg * 32];
        unsigned long long ga = (unsigned long long)(uintptr_t)s;
        asm volatile(
            "global_load_async_to_lds_b128 %0, %1, off\n\t"
            "global_load_async_to_lds_b128 %0, %1, off offset:16\n\t"
            "global_load_async_to_lds_b128 %0, %1, off offset:32\n\t"
            "global_load_async_to_lds_b128 %0, %1, off offset:48"
            :: "v"(dst), "v"(ga) : "memory");
    };

    stage(0, 0);
    asm volatile("s_wait_asynccnt 0x0" ::: "memory");
    __syncthreads();

    const int r16 = lane & 15;
    const int kb8 = lane >> 4;

    for (int kcw = 0; kcw < KCW; ++kcw) {
        int buf = kcw & 1;
        if (kcw + 1 < KCW) stage(kcw + 1, buf ^ 1);
        // prefetch the B-weight stream ~8 x 32-chunks (8KB/lane-group) ahead
        __builtin_prefetch(bw + (size_t)(kcw * 2 + 8) * 512, 0, 3);

#pragma unroll
        for (int h = 0; h < 2; ++h) {
            FragU fb;
            const uint4* bp = (const uint4*)(bw + (size_t)(kcw * 2 + h) * 512);
            fb.q[0] = bp[0]; fb.q[1] = bp[1];
#pragma unroll
            for (int i = 0; i < 4; ++i) {
                FragU fa;
                const uint4* ap =
                    (const uint4*)&As[buf][mhalf * 64 + i * 16 + r16][0];
                fa.q[0] = ap[h * 4 + kb8];
                fa.q[1] = ap[h * 4 + 2 + kb8];
                acc[i] = __builtin_amdgcn_wmma_f32_16x16x32_bf16(
                    false, fa.v, false, fb.v, (short)0, acc[i], false, false);
            }
        }
        asm volatile("s_wait_asynccnt 0x0" ::: "memory");
        __syncthreads();
    }

    // C/D layout: VGPR v -> row v + 8*(lane>>4), col = lane&15 (ISA 7.12.2)
#pragma unroll
    for (int i = 0; i < 4; ++i) {
        int rbase = mhalf * 64 + i * 16 + ((lane >> 4) << 3);
        int col   = ntile * 16 + (lane & 15);
#pragma unroll
        for (int v = 0; v < 8; ++v) Zs[rbase + v][col] = acc[i][v];
    }
    __syncthreads();

    // LSTM gate epilogue: cols 0..15=i, 16..31=f, 32..47=g, 48..63=o
    for (int idx = tid; idx < BATCH * 16; idx += 256) {
        int row = idx >> 4, ul = idx & 15;
        int u = wg * 16 + ul;                       // global unit index
        float zi = Zs[row][ul]      + bias[u];
        float zf = Zs[row][16 + ul] + bias[2048 + u];
        float zg = Zs[row][32 + ul] + bias[4096 + u];
        float zo = Zs[row][48 + ul] + bias[6144 + u];
        size_t off = (size_t)row * UNITS + u;
        float cn = sigm(zf) * c[off] + sigm(zi) * tanhf(zg);
        float hn = sigm(zo) * tanhf(cn);
        c[off]    = cn;
        hout[off] = f2bf(hn);
        if (hf) hf[off] = hn;
    }
}

// ---------------------------- output head ----------------------------------
__global__ __launch_bounds__(256) void k_dense(const float* __restrict__ h1f,
                                               const float* __restrict__ Wd,
                                               const float* __restrict__ bd,
                                               float* __restrict__ y) {
    size_t idx = (size_t)blockIdx.x * 256 + threadIdx.x;  // 128*2048
    int row = (int)(idx >> 11), col = (int)(idx & 2047);
    float s = bd[col];
    const float* hr = h1f + (size_t)row * UNITS;
    for (int k = 0; k < UNITS; ++k) s += hr[k] * Wd[(size_t)k * UNITS + col];
    y[idx] = fmaxf(s, 0.0f);
}

__global__ __launch_bounds__(256) void k_out(const float* __restrict__ y,
                                             const float* __restrict__ Wo,
                                             const float* __restrict__ bo,
                                             float* __restrict__ out) {
    __shared__ float red[256];
    int row = blockIdx.x, tid = threadIdx.x;
    float s = 0.0f;
    const float* yr = y + (size_t)row * UNITS;
    for (int k = tid; k < UNITS; k += 256) s += yr[k] * Wo[k];
    red[tid] = s;
    __syncthreads();
    for (int w = 128; w > 0; w >>= 1) {
        if (tid < w) red[tid] += red[tid + w];
        __syncthreads();
    }
    if (tid == 0) out[row] = sigm(red[0] + bo[0]);
}

// ---------------------------------------------------------------------------
extern "C" void kernel_launch(void* const* d_in, const int* in_sizes, int n_in,
                              void* d_out, int out_size, void* d_ws, size_t ws_size,
                              hipStream_t stream) {
    const int*   tokens = (const int*)  d_in[0];
    const float* emb    = (const float*)d_in[1];
    const float* W0     = (const float*)d_in[2];
    const float* U0     = (const float*)d_in[3];
    const float* b0     = (const float*)d_in[4];
    const float* W1     = (const float*)d_in[5];
    const float* U1     = (const float*)d_in[6];
    const float* b1     = (const float*)d_in[7];
    const float* Wd     = (const float*)d_in[8];
    const float* bd     = (const float*)d_in[9];
    const float* Wo     = (const float*)d_in[10];
    const float* bo     = (const float*)d_in[11];
    float* out = (float*)d_out;

    char* ws = (char*)d_ws;
    size_t off = 0;
    // --- zero-initialized state block (contiguous) ---
    float*          c0   = (float*)(ws + off);          off += (size_t)BATCH * UNITS * 4;   // 1 MB
    float*          c1   = (float*)(ws + off);          off += (size_t)BATCH * UNITS * 4;   // 1 MB
    float*          h1f  = (float*)(ws + off);          off += (size_t)BATCH * UNITS * 4;   // 1 MB
    unsigned short* h0bf = (unsigned short*)(ws + off); off += (size_t)2 * BATCH * UNITS * 2; // 1 MB
    unsigned short* h1bf = (unsigned short*)(ws + off); off += (size_t)2 * BATCH * UNITS * 2; // 1 MB
    size_t stateBytes = off;                                                               // 5 MB
    // --- constants rebuilt per launch ---
    unsigned short* wu0 = (unsigned short*)(ws + off);  off += (size_t)NTILES * KC0 * 512 * 2; // 35.7 MB
    unsigned short* wu1 = (unsigned short*)(ws + off);  off += (size_t)NTILES * KC1 * 512 * 2; // 67.1 MB
    unsigned short* xg  = (unsigned short*)(ws + off);  off += (size_t)SEQT * BATCH * EPAD * 2; // 2.6 MB
    float*          y   = (float*)(ws + off);           off += (size_t)BATCH * UNITS * 4;   // 1 MB
    (void)ws_size; (void)n_in; (void)in_sizes; (void)out_size;   // total ~112 MB

    // 1) zero states
    {
        size_t n = stateBytes / 4;
        k_zero<<<(unsigned)((n + 255) / 256), 256, 0, stream>>>((unsigned int*)ws, n);
    }
    // 2) weight swizzle: [W0(pad 100->128); U0] and [W1; U1]
    {
        size_t t0 = (size_t)NTILES * KC0 * 512;
        k_swizzle<<<(unsigned)((t0 + 255) / 256), 256, 0, stream>>>(
            W0, EMB, EPAD, U0, wu0, KC0);
        size_t t1 = (size_t)NTILES * KC1 * 512;
        k_swizzle<<<(unsigned)((t1 + 255) / 256), 256, 0, stream>>>(
            W1, UNITS, UNITS, U1, wu1, KC1);
    }
    // 3) embedding gather
    {
        size_t n = (size_t)SEQT * BATCH * EPAD;
        k_embed<<<(unsigned)((n + 255) / 256), 256, 0, stream>>>(tokens, emb, xg);
    }
    // 4) recurrent scan: 2 fused GEMM+gate kernels per timestep
    const size_t HSZ = (size_t)BATCH * UNITS;  // elements per h buffer
    for (int t = 0; t < SEQT; ++t) {
        int cur = t & 1, nxt = cur ^ 1;
        // layer 0: z = [x_t | h0] @ [W0;U0] + b0
        k_lstm_step<<<128, 256, 0, stream>>>(
            wu0, KC0,
            xg + (size_t)t * BATCH * EPAD, EPAD, EPAD,
            h0bf + (size_t)cur * HSZ, UNITS,
            b0, c0, h0bf + (size_t)nxt * HSZ, nullptr);
        // layer 1: z = [h0' | h1] @ [W1;U1] + b1
        k_lstm_step<<<128, 256, 0, stream>>>(
            wu1, KC1,
            h0bf + (size_t)nxt * HSZ, UNITS, UNITS,
            h1bf + (size_t)cur * HSZ, UNITS,
            b1, c1, h1bf + (size_t)nxt * HSZ, h1f);
    }
    // 5) head: Dense -> ReLU -> Dense(1) -> sigmoid
    k_dense<<<(unsigned)(((size_t)BATCH * UNITS + 255) / 256), 256, 0, stream>>>(
        h1f, Wd, bd, y);
    k_out<<<BATCH, 256, 0, stream>>>(y, Wo, bo, out);
}
